// DirectionalBiCMS_70738111365378
// MI455X (gfx1250) — matrix operand order repaired and checked by reading, not compiled, not run on hardware
//
#include <hip/hip_runtime.h>

// ---------------- CDNA5 WMMA types ----------------
typedef __attribute__((ext_vector_type(16))) __bf16 v16bf;
typedef __attribute__((ext_vector_type(8)))  __bf16 v8bf;
typedef __attribute__((ext_vector_type(8)))  float  v8f;
typedef __attribute__((ext_vector_type(4)))  float  v4f;

// Model constants
#define BSZ   8
#define LSEQ  4096            // 2 * L_SCAN (interleaved length)
#define DMODEL 256
#define DINNER 512
#define CONVD  640
#define HEADS  8
#define NSTATE 64
#define PDIM   64
#define NPROJ  1160           // 512 + 640 + 8
#define TTOK   (BSZ * LSEQ)   // 32768 tokens
#define NCHUNK 64             // LSEQ / 64 chunks of Q=64
#define TSTR   72             // LDS tile row stride (bf16), 16B-aligned rows

static __device__ __forceinline__ v16bf mk16(v8bf lo, v8bf hi) {
  v16bf f;
  #pragma unroll
  for (int i = 0; i < 8; ++i) { f[i] = lo[i]; f[8 + i] = hi[i]; }
  return f;
}

// 64x64x64 bf16 WMMA macro-tile: 8 waves, wave = (mt = w&3, nt-pair = (w>>2)*2).
static __device__ __forceinline__ void wmma64(const __bf16* At, const __bf16* Bt,
                                              int lane, int mt, int nt,
                                              v8f& acc0, v8f& acc1) {
  const int am   = mt * 16 + (lane & 15);
  const int bn0  = nt * 16 + (lane & 15);
  const int asel = (lane >> 4) * 8;
  const int bsel = (lane >> 4) * 8;
  #pragma unroll
  for (int ks = 0; ks < 64; ks += 32) {
    const __bf16* ar = At + am * TSTR + ks;
    v16bf af = mk16(*(const v8bf*)(ar + asel), *(const v8bf*)(ar + asel + 16));
    const __bf16* br0 = Bt + bn0 * TSTR + ks + bsel;
    const __bf16* br1 = Bt + (bn0 + 16) * TSTR + ks + bsel;
    v16bf bf0 = mk16(*(const v8bf*)br0, *(const v8bf*)(br0 + 16));
    v16bf bf1 = mk16(*(const v8bf*)br1, *(const v8bf*)(br1 + 16));
    acc0 = __builtin_amdgcn_wmma_f32_16x16x32_bf16(false, af, false, bf0,
                                                   (short)0, acc0, false, false);
    acc1 = __builtin_amdgcn_wmma_f32_16x16x32_bf16(false, af, false, bf1,
                                                   (short)0, acc1, false, false);
  }
}

// =====================================================================
// One-time f32 -> bf16 conversion (weights).
// =====================================================================
__global__ void f32_to_bf16(const float* __restrict__ src, __bf16* __restrict__ dst, int n)
{
  int i = blockIdx.x * blockDim.x + threadIdx.x;
  if (i < n) dst[i] = (__bf16)src[i];
}

// =====================================================================
// GEMM: out(MxN) = A(MxK)bf16 @ W(KxN)bf16 [+ bias] [+ res], f32 out.
// 256 threads / 8 waves, 64x64 tile, double-buffered LDS.
// A tile staged with GLOBAL_LOAD_ASYNC_TO_LDS_B128 (ASYNCcnt);
// W tile staged synchronously with a bf16-pair transpose repack.
// =====================================================================
__global__ __launch_bounds__(256) void gemm_bf16_wmma(
    const __bf16* __restrict__ A, const __bf16* __restrict__ W,
    const float* __restrict__ bias, const float* __restrict__ res,
    float* __restrict__ out, int M, int N, int K)
{
  __shared__ __align__(16) __bf16 As[2][64][40];
  __shared__ __align__(16) __bf16 Bs[2][64][40];   // W tile transposed: [n][k]

  const int tid  = threadIdx.x;
  const int lane = tid & 31;
  const int wave = tid >> 5;
  const int mt   = wave & 3;
  const int nt   = (wave >> 2) * 2;
  const int row0 = blockIdx.x * 64;
  const int col0 = blockIdx.y * 64;

  v8f acc0 = {}; v8f acc1 = {};

  const int ar  = tid >> 2;           // A row 0..63
  const int akk = (tid & 3) * 8;      // A k chunk (8 bf16 = 16B)
  const int wkk = (tid & 15) * 2;     // W k pair
  const int wnn = (tid >> 4) * 4;     // W 4-col chunk
  const bool wvalid = (col0 + wnn < N);

  auto stage = [&](int k0, int buf) {
    // ---- A: async global -> LDS, one b128 per thread ----
    unsigned lds = (unsigned)(size_t)&As[buf][ar][akk];
    unsigned long long ga =
        (unsigned long long)(size_t)(A + (size_t)(row0 + ar) * K + k0 + akk);
    asm volatile("global_load_async_to_lds_b128 %0, %1, off"
                 :: "v"(lds), "v"(ga) : "memory");
    // ---- W: transpose repack (2 k-rows x 4 cols -> 4 packed pair stores) ----
    if (wvalid) {
      const __bf16* srcW = W + (size_t)(k0 + wkk) * N + col0 + wnn;
      unsigned long long r0 = *(const unsigned long long*)srcW;
      unsigned long long r1 = *(const unsigned long long*)(srcW + N);
      #pragma unroll
      for (int j = 0; j < 4; ++j) {
        unsigned lo = (unsigned)((r0 >> (16 * j)) & 0xffffu);
        unsigned hi = (unsigned)((r1 >> (16 * j)) & 0xffffu);
        *(unsigned*)&Bs[buf][wnn + j][wkk] = lo | (hi << 16);
      }
    } else {
      #pragma unroll
      for (int j = 0; j < 4; ++j) *(unsigned*)&Bs[buf][wnn + j][wkk] = 0u;
    }
  };

  stage(0, 0);
  int buf = 0;
  for (int k0 = 0; k0 < K; k0 += 32, buf ^= 1) {
    asm volatile("s_wait_asynccnt 0x0" ::: "memory");  // our async A landed
    __syncthreads();                                    // all waves' tile ready
    if (k0 + 32 < K) stage(k0 + 32, buf ^ 1);           // prefetch next tile

    const int am = mt * 16 + (lane & 15);
    const int ak = (lane >> 4) * 8;
    v16bf afrag = mk16(*(const v8bf*)&As[buf][am][ak],
                       *(const v8bf*)&As[buf][am][ak + 16]);
    const int bn0 = nt * 16 + (lane & 15);
    const int bkk = (lane >> 4) * 8;
    v16bf bfrag0 = mk16(*(const v8bf*)&Bs[buf][bn0][bkk],
                        *(const v8bf*)&Bs[buf][bn0][bkk + 16]);
    v16bf bfrag1 = mk16(*(const v8bf*)&Bs[buf][bn0 + 16][bkk],
                        *(const v8bf*)&Bs[buf][bn0 + 16][bkk + 16]);

    acc0 = __builtin_amdgcn_wmma_f32_16x16x32_bf16(false, afrag, false, bfrag0,
                                                   (short)0, acc0, false, false);
    acc1 = __builtin_amdgcn_wmma_f32_16x16x32_bf16(false, afrag, false, bfrag1,
                                                   (short)0, acc1, false, false);
  }

  const int mrow = row0 + mt * 16 + ((lane >> 4) ? 8 : 0);
  const int c0 = col0 + nt * 16 + (lane & 15);
  const int c1 = c0 + 16;
  #pragma unroll
  for (int r = 0; r < 8; ++r) {
    const int m = mrow + r;
    if (c0 < N) {
      float v = acc0[r];
      if (bias) v += bias[c0];
      if (res)  v += res[(size_t)m * N + c0];
      out[(size_t)m * N + c0] = v;
    }
    if (c1 < N) {
      float v = acc1[r];
      if (bias) v += bias[c1];
      if (res)  v += res[(size_t)m * N + c1];
      out[(size_t)m * N + c1] = v;
    }
  }
}

// =====================================================================
// Interleave a0,b0,a1,b1,... -> fwd activation; reversed copy -> bwd.
// =====================================================================
__global__ void interleave_kernel(const float* __restrict__ sa, const float* __restrict__ sb,
                                  float* __restrict__ fwd, float* __restrict__ bwd)
{
  int idx = blockIdx.x * blockDim.x + threadIdx.x;
  if (idx >= TTOK * DMODEL) return;
  int d  = idx % DMODEL;
  int t2 = (idx / DMODEL) % LSEQ;
  int b  = idx / (DMODEL * LSEQ);
  const float* src = (t2 & 1) ? sb : sa;
  float v = src[((size_t)b * (LSEQ / 2) + (t2 >> 1)) * DMODEL + d];
  fwd[idx] = v;
  bwd[((size_t)b * LSEQ + (LSEQ - 1 - t2)) * DMODEL + d] = v;
}

// =====================================================================
// RMSNorm over D=256 (one wave per token) -> bf16 output for the GEMM.
// =====================================================================
__global__ __launch_bounds__(256) void rmsnorm256(const float* __restrict__ x,
                                                  const float* __restrict__ w,
                                                  __bf16* __restrict__ o, int M)
{
  int wave = threadIdx.x >> 5, lane = threadIdx.x & 31;
  int row = blockIdx.x * 8 + wave;
  if (row >= M) return;
  const float* xr = x + (size_t)row * DMODEL;
  float v[8]; float ss = 0.f;
  #pragma unroll
  for (int j = 0; j < 8; ++j) { v[j] = xr[lane + j * 32]; ss += v[j] * v[j]; }
  #pragma unroll
  for (int m = 16; m >= 1; m >>= 1) ss += __shfl_xor(ss, m, 32);
  float sc = rsqrtf(ss * (1.f / 256.f) + 1e-6f);
  __bf16* orow = o + (size_t)row * DMODEL;
  #pragma unroll
  for (int j = 0; j < 8; ++j)
    orow[lane + j * 32] = (__bf16)(v[j] * sc * w[lane + j * 32]);
}

// =====================================================================
// Depthwise causal conv (K=4) + SiLU on xBC slice of proj.
// =====================================================================
__global__ void conv_silu(const float* __restrict__ proj, const float* __restrict__ cw,
                          const float* __restrict__ cb, float* __restrict__ xbc)
{
  int idx = blockIdx.x * blockDim.x + threadIdx.x;
  if (idx >= TTOK * CONVD) return;
  int c  = idx % CONVD;
  int tk = idx / CONVD;
  int t  = tk % LSEQ;
  float acc = cb[c];
  #pragma unroll
  for (int k = 0; k < 4; ++k) {
    int ts = t - 3 + k;
    if (ts >= 0) acc += cw[c * 4 + k] * proj[(size_t)(tk - 3 + k) * NPROJ + DINNER + c];
  }
  xbc[idx] = acc / (1.f + __expf(-acc));
}

// =====================================================================
// dt = softplus(proj[...,1152:1160] + dt_bias)
// =====================================================================
__global__ void dt_softplus(const float* __restrict__ proj, const float* __restrict__ dt_bias,
                            float* __restrict__ dt)
{
  int idx = blockIdx.x * blockDim.x + threadIdx.x;
  if (idx >= TTOK * HEADS) return;
  int h = idx & 7;
  float x = proj[(size_t)(idx >> 3) * NPROJ + (DINNER + CONVD) + h] + dt_bias[h];
  dt[idx] = (x > 20.f) ? x : log1pf(__expf(x));
}

// =====================================================================
// lam[tk,h] = inclusive cumsum (within 64-chunk) of dt * A_h, A_h = -exp(A_log).
// =====================================================================
__global__ void lam_cumsum(const float* __restrict__ dt, const float* __restrict__ A_log,
                           float* __restrict__ lam)
{
  int r = blockIdx.x * blockDim.x + threadIdx.x;
  if (r >= BSZ * NCHUNK * HEADS) return;
  int h = r & 7, c = (r >> 3) & 63, b = r >> 9;
  const float Ah = -__expf(A_log[h]);
  size_t base = ((size_t)b * LSEQ + c * 64) * HEADS + h;
  float run = 0.f;
  for (int i = 0; i < 64; ++i) {
    run += dt[base + (size_t)i * HEADS] * Ah;
    lam[base + (size_t)i * HEADS] = run;
  }
}

// =====================================================================
// Chunked SSD, intra-chunk (one workgroup per (b,chunk,h)):
//   G = C@B^T (WMMA); M = causal-decay mask(G); Yintra = M@X (+D*x);
//   S = Bscaled^T @ X.
// =====================================================================
__global__ __launch_bounds__(256) void ssd_chunk_intra(
    const float* __restrict__ xbc, const float* __restrict__ dt,
    const float* __restrict__ lam, const float* __restrict__ Dp,
    float* __restrict__ y, float* __restrict__ Sbuf)
{
  const int blk = blockIdx.x;
  const int h = blk & 7;
  const int c = (blk >> 3) & 63;
  const int b = blk >> 9;
  const int tk0 = b * LSEQ + c * 64;
  __shared__ __align__(16) __bf16 Ct[64 * TSTR];    // [i][n]
  __shared__ __align__(16) __bf16 Bt[64 * TSTR];    // [s][n]
  __shared__ __align__(16) __bf16 Bsc[64 * TSTR];   // [n][s] scaled transpose
  __shared__ __align__(16) __bf16 Xt[64 * TSTR];    // [p][s] transpose
  __shared__ __align__(16) __bf16 Mt[64 * TSTR];    // [i][s] masked scores
  __shared__ float sLam[64], sdt[64];
  const int tid = threadIdx.x, lane = tid & 31, wave = tid >> 5;
  const int mt = wave & 3, nt = (wave >> 2) * 2;

  if (tid < 64) {
    sLam[tid] = lam[(size_t)(tk0 + tid) * HEADS + h];
    sdt[tid]  = dt[(size_t)(tk0 + tid) * HEADS + h];
  }
  __syncthreads();
  const float lamEnd = sLam[63];

  {
    const int s = tid >> 2;
    const int q = (tid & 3) * 16;
    const float* row = xbc + (size_t)(tk0 + s) * CONVD;
    const float ws = __expf(lamEnd - sLam[s]) * sdt[s];
    #pragma unroll
    for (int j = 0; j < 16; j += 4) {
      v4f bv = *(const v4f*)(row + DINNER + q + j);
      v4f cv = *(const v4f*)(row + DINNER + NSTATE + q + j);
      v4f xv = *(const v4f*)(row + h * PDIM + q + j);
      #pragma unroll
      for (int e = 0; e < 4; ++e) {
        int n = q + j + e;
        Bt[s * TSTR + n]  = (__bf16)bv[e];
        Ct[s * TSTR + n]  = (__bf16)cv[e];
        Bsc[n * TSTR + s] = (__bf16)(bv[e] * ws);
        Xt[n * TSTR + s]  = (__bf16)xv[e];
      }
    }
  }
  __syncthreads();

  v8f g0 = {}, g1 = {};
  wmma64(Ct, Bt, lane, mt, nt, g0, g1);

  {
    const int mrow = mt * 16 + ((lane >> 4) ? 8 : 0);
    const int c0 = nt * 16 + (lane & 15);
    #pragma unroll
    for (int r = 0; r < 8; ++r) {
      int i = mrow + r;
      float m0 = (c0 <= i)      ? g0[r] * __expf(sLam[i] - sLam[c0])      * sdt[c0]      : 0.f;
      float m1 = (c0 + 16 <= i) ? g1[r] * __expf(sLam[i] - sLam[c0 + 16]) * sdt[c0 + 16] : 0.f;
      Mt[i * TSTR + c0]      = (__bf16)m0;
      Mt[i * TSTR + c0 + 16] = (__bf16)m1;
    }
  }
  __syncthreads();

  v8f y0 = {}, y1 = {};
  wmma64(Mt, Xt, lane, mt, nt, y0, y1);
  v8f s0 = {}, s1 = {};
  wmma64(Bsc, Xt, lane, mt, nt, s0, s1);

  const float Dh = Dp[h];
  const int mrow = mt * 16 + ((lane >> 4) ? 8 : 0);
  const int c0 = nt * 16 + (lane & 15);
  float* Sout = Sbuf + (((size_t)(b * HEADS + h)) * NCHUNK + c) * 4096;
  #pragma unroll
  for (int r = 0; r < 8; ++r) {
    int i = mrow + r;
    float x0 = (float)Xt[c0 * TSTR + i];
    float x1 = (float)Xt[(c0 + 16) * TSTR + i];
    y[(size_t)(tk0 + i) * DINNER + h * PDIM + c0]      = y0[r] + Dh * x0;
    y[(size_t)(tk0 + i) * DINNER + h * PDIM + c0 + 16] = y1[r] + Dh * x1;
    Sout[i * 64 + c0]      = s0[r];
    Sout[i * 64 + c0 + 16] = s1[r];
  }
}

// =====================================================================
// Serial inter-chunk carry (64 steps) per (b,h).
// =====================================================================
__global__ __launch_bounds__(256) void ssd_carry(
    const float* __restrict__ lam, float* __restrict__ Sbuf)
{
  const int b = blockIdx.x >> 3, h = blockIdx.x & 7;
  const int tid = threadIdx.x;
  float R[16];
  #pragma unroll
  for (int j = 0; j < 16; ++j) R[j] = 0.f;
  float* Sb = Sbuf + ((size_t)(b * HEADS + h)) * NCHUNK * 4096;
  for (int c = 0; c < NCHUNK; ++c) {
    float dec = __expf(lam[(size_t)(b * LSEQ + c * 64 + 63) * HEADS + h]);
    float* Sc = Sb + (size_t)c * 4096;
    #pragma unroll
    for (int j = 0; j < 16; ++j) {
      int e = tid * 16 + j;
      float s = Sc[e];
      Sc[e] = R[j];
      R[j] = dec * R[j] + s;
    }
  }
}

// =====================================================================
// Chunked SSD, inter-chunk: y += (exp(Lam_i) * C_i) @ R_prev.
// =====================================================================
__global__ __launch_bounds__(256) void ssd_chunk_inter(
    const float* __restrict__ xbc, const float* __restrict__ lam,
    const float* __restrict__ Sbuf, float* __restrict__ y)
{
  const int blk = blockIdx.x;
  const int h = blk & 7;
  const int c = (blk >> 3) & 63;
  const int b = blk >> 9;
  const int tk0 = b * LSEQ + c * 64;
  __shared__ __align__(16) __bf16 Csc[64 * TSTR];  // [i][n], row-scaled
  __shared__ __align__(16) __bf16 Rt[64 * TSTR];   // [p][n]
  __shared__ float sLam[64];
  const int tid = threadIdx.x, lane = tid & 31, wave = tid >> 5;
  const int mt = wave & 3, nt = (wave >> 2) * 2;

  if (tid < 64) sLam[tid] = lam[(size_t)(tk0 + tid) * HEADS + h];
  __syncthreads();

  {
    const int s = tid >> 2;
    const int q = (tid & 3) * 16;
    const float sc = __expf(sLam[s]);
    const float* crow = xbc + (size_t)(tk0 + s) * CONVD + DINNER + NSTATE;
    const float* rrow = Sbuf + (((size_t)(b * HEADS + h)) * NCHUNK + c) * 4096 + s * 64;
    #pragma unroll
    for (int j = 0; j < 16; j += 4) {
      v4f cv = *(const v4f*)(crow + q + j);
      v4f rv = *(const v4f*)(rrow + q + j);
      #pragma unroll
      for (int e = 0; e < 4; ++e) {
        int n = q + j + e;
        Csc[s * TSTR + n] = (__bf16)(cv[e] * sc);
        Rt[n * TSTR + s]  = (__bf16)rv[e];
      }
    }
  }
  __syncthreads();

  v8f a0 = {}, a1 = {};
  wmma64(Csc, Rt, lane, mt, nt, a0, a1);

  const int mrow = mt * 16 + ((lane >> 4) ? 8 : 0);
  const int c0 = nt * 16 + (lane & 15);
  #pragma unroll
  for (int r = 0; r < 8; ++r) {
    int i = mrow + r;
    size_t o0 = (size_t)(tk0 + i) * DINNER + h * PDIM + c0;
    y[o0]      += a0[r];
    y[o0 + 16] += a1[r];
  }
}

// =====================================================================
// Gated RMSNorm over D_INNER=512 -> bf16 output for out_proj GEMM.
// =====================================================================
__global__ __launch_bounds__(256) void gated_rmsnorm512(
    const float* __restrict__ y, const float* __restrict__ proj,
    const float* __restrict__ w, __bf16* __restrict__ obf, int M)
{
  int wave = threadIdx.x >> 5, lane = threadIdx.x & 31;
  int row = blockIdx.x * 8 + wave;
  if (row >= M) return;
  const float* yr = y + (size_t)row * DINNER;
  const float* zr = proj + (size_t)row * NPROJ;
  float v[16]; float ss = 0.f;
  #pragma unroll
  for (int j = 0; j < 16; ++j) {
    int i = lane + j * 32;
    float z = zr[i];
    float val = yr[i] * (z / (1.f + __expf(-z)));
    v[j] = val; ss += val * val;
  }
  #pragma unroll
  for (int m = 16; m >= 1; m >>= 1) ss += __shfl_xor(ss, m, 32);
  float sc = rsqrtf(ss * (1.f / 512.f) + 1e-6f);
  __bf16* orow = obf + (size_t)row * DINNER;
  #pragma unroll
  for (int j = 0; j < 16; ++j) {
    int i = lane + j * 32;
    orow[i] = (__bf16)(v[j] * sc * w[i]);
  }
}

// =====================================================================
// Build concat(fwd, flip(bwd)) in bf16 for the gate GEMM.
// =====================================================================
__global__ void build_concat(const float* __restrict__ f, const float* __restrict__ wrev,
                             __bf16* __restrict__ cat)
{
  int idx = blockIdx.x * blockDim.x + threadIdx.x;
  if (idx >= TTOK * 512) return;
  int c  = idx % 512;
  int tk = idx / 512;
  int t = tk % LSEQ, b = tk / LSEQ;
  float v = (c < DMODEL)
      ? f[(size_t)tk * DMODEL + c]
      : wrev[((size_t)b * LSEQ + (LSEQ - 1 - t)) * DMODEL + (c - DMODEL)];
  cat[idx] = (__bf16)v;
}

// =====================================================================
// Sigmoid-gated merge + de-interleave into the two outputs.
// =====================================================================
__global__ void merge_out(const float* __restrict__ f, const float* __restrict__ wrev,
                          const float* __restrict__ glog, float* __restrict__ outp)
{
  int idx = blockIdx.x * blockDim.x + threadIdx.x;
  if (idx >= TTOK * DMODEL) return;
  int d  = idx % DMODEL;
  int tk = idx / DMODEL;
  int t = tk % LSEQ, b = tk / LSEQ;
  float g  = 1.f / (1.f + __expf(-glog[idx]));
  float fv = f[idx];
  float bv = wrev[((size_t)b * LSEQ + (LSEQ - 1 - t)) * DMODEL + d];
  float m  = g * fv + (1.f - g) * bv;
  const size_t half = (size_t)TTOK * DMODEL / 2;
  size_t o = ((size_t)b * (LSEQ / 2) + (t >> 1)) * DMODEL + d;
  outp[(t & 1) ? (half + o) : o] = m;
}

// =====================================================================
// Host launch. d_in order: 0 stream_a, 1 stream_b, 2..19 fwd_params
// (2 layers x 9: norm_w, in_proj_W, conv_w, conv_b, dt_bias, A_log, D,
// gated_norm_w, out_proj_W), 20..37 bwd_params, 38 gate_W, 39 gate_b.
// =====================================================================
extern "C" void kernel_launch(void* const* d_in, const int* in_sizes, int n_in,
                              void* d_out, int out_size, void* d_ws, size_t ws_size,
                              hipStream_t stream)
{
  (void)in_sizes; (void)n_in; (void)out_size; (void)ws_size;
  const float* stream_a = (const float*)d_in[0];
  const float* stream_b = (const float*)d_in[1];
  const float* gate_W = (const float*)d_in[38];
  const float* gate_b = (const float*)d_in[39];

  float* ws = (float*)d_ws;
  size_t o = 0;
  float* runF = ws + o;  o += (size_t)TTOK * DMODEL;
  float* runB = ws + o;  o += (size_t)TTOK * DMODEL;
  float* proj = ws + o;  o += (size_t)TTOK * NPROJ;
  float* xbc  = ws + o;  o += (size_t)TTOK * CONVD;
  float* dtb  = ws + o;  o += (size_t)TTOK * HEADS;
  float* yb   = ws + o;  o += (size_t)TTOK * DINNER;
  float* lamb = ws + o;  o += (size_t)TTOK * HEADS;
  float* Sbuf = ws + o;  o += (size_t)BSZ * HEADS * NCHUNK * 4096;
  float* glog = xbc;    // reused after both stacks complete

  // bf16 region (carved after the f32 region; offsets stay 16B aligned)
  __bf16* bh = (__bf16*)(ws + o);
  size_t ob = 0;
  __bf16* xnbf = bh + ob;  ob += (size_t)TTOK * DMODEL;
  __bf16* ybf  = bh + ob;  ob += (size_t)TTOK * DINNER;
  __bf16* catbf = ybf;     // gate input reuses ybf region after stacks
  __bf16* wInBf[4];
  for (int i = 0; i < 4; ++i) { wInBf[i] = bh + ob; ob += (size_t)DMODEL * NPROJ; }
  __bf16* wOutBf[4];
  for (int i = 0; i < 4; ++i) { wOutBf[i] = bh + ob; ob += (size_t)DINNER * DMODEL; }
  __bf16* wGateBf = bh + ob; ob += (size_t)DINNER * DMODEL;

  const dim3 blk(256);

  // ---- one-time weight conversions (f32 -> bf16) ----
  for (int dir = 0; dir < 2; ++dir)
    for (int layer = 0; layer < 2; ++layer) {
      int idx = dir * 2 + layer;
      const float* inW  = (const float*)d_in[2 + dir * 18 + layer * 9 + 1];
      const float* outW = (const float*)d_in[2 + dir * 18 + layer * 9 + 8];
      f32_to_bf16<<<(DMODEL * NPROJ + 255) / 256, blk, 0, stream>>>(inW, wInBf[idx], DMODEL * NPROJ);
      f32_to_bf16<<<(DINNER * DMODEL + 255) / 256, blk, 0, stream>>>(outW, wOutBf[idx], DINNER * DMODEL);
    }
  f32_to_bf16<<<(DINNER * DMODEL + 255) / 256, blk, 0, stream>>>(gate_W, wGateBf, DINNER * DMODEL);

  interleave_kernel<<<(TTOK * DMODEL + 255) / 256, blk, 0, stream>>>(stream_a, stream_b, runF, runB);

  for (int dir = 0; dir < 2; ++dir) {
    float* run = dir ? runB : runF;
    for (int layer = 0; layer < 2; ++layer) {
      int idx = dir * 2 + layer;
      const float* const* prm = (const float* const*)&d_in[2 + dir * 18 + layer * 9];
      const float* norm_w  = prm[0];
      const float* conv_w  = prm[2];
      const float* conv_b  = prm[3];
      const float* dt_bias = prm[4];
      const float* A_log   = prm[5];
      const float* Dp      = prm[6];
      const float* gn_w    = prm[7];

      rmsnorm256<<<TTOK / 8, blk, 0, stream>>>(run, norm_w, xnbf, TTOK);
      {
        dim3 g(TTOK / 64, (NPROJ + 63) / 64);
        gemm_bf16_wmma<<<g, blk, 0, stream>>>(xnbf, wInBf[idx], nullptr, nullptr, proj,
                                              TTOK, NPROJ, DMODEL);
      }
      conv_silu<<<(TTOK * CONVD + 255) / 256, blk, 0, stream>>>(proj, conv_w, conv_b, xbc);
      dt_softplus<<<(TTOK * HEADS + 255) / 256, blk, 0, stream>>>(proj, dt_bias, dtb);
      lam_cumsum<<<(BSZ * NCHUNK * HEADS + 255) / 256, blk, 0, stream>>>(dtb, A_log, lamb);
      ssd_chunk_intra<<<BSZ * NCHUNK * HEADS, blk, 0, stream>>>(xbc, dtb, lamb, Dp, yb, Sbuf);
      ssd_carry<<<BSZ * HEADS, blk, 0, stream>>>(lamb, Sbuf);
      ssd_chunk_inter<<<BSZ * NCHUNK * HEADS, blk, 0, stream>>>(xbc, lamb, Sbuf, yb);
      gated_rmsnorm512<<<TTOK / 8, blk, 0, stream>>>(yb, proj, gn_w, ybf, TTOK);
      {
        dim3 g(TTOK / 64, DMODEL / 64);
        gemm_bf16_wmma<<<g, blk, 0, stream>>>(ybf, wOutBf[idx], nullptr, run, run,
                                              TTOK, DMODEL, DINNER);
      }
    }
  }

  build_concat<<<(TTOK * 512 + 255) / 256, blk, 0, stream>>>(runF, runB, catbf);
  {
    dim3 g(TTOK / 64, DMODEL / 64);
    gemm_bf16_wmma<<<g, blk, 0, stream>>>(catbf, wGateBf, gate_b, nullptr, glog,
                                          TTOK, DMODEL, DINNER);
  }
  merge_out<<<(TTOK * DMODEL + 255) / 256, blk, 0, stream>>>(runF, runB, glog, (float*)d_out);
}
